// MultiHeadAttentionFromScratch_10462540333416
// MI455X (gfx1250) — matrix operand order repaired and checked by reading, not compiled, not run on hardware
//
#include <hip/hip_runtime.h>
#include <hip/hip_bf16.h>

// MI455X (gfx1250) multi-head attention:
//   prep (fp32->bf16 + weight transpose) -> QKV GEMM (WMMA bf16) ->
//   flash attention (WMMA bf16, online softmax) -> output GEMM (WMMA bf16, f32 out)
// Matrix math on v_wmma_f32_16x16x32_bf16 (wave32, f32 accum).
// LDS staging uses GLOBAL_LOAD_ASYNC_TO_LDS_B128 (ASYNCcnt) when the toolchain
// exposes the builtin; otherwise batched register staging.

typedef __bf16 bf16_t;
typedef __attribute__((ext_vector_type(16))) __bf16 v16bf;
typedef __attribute__((ext_vector_type(8)))  __bf16 v8bf;
typedef __attribute__((ext_vector_type(8)))  float  v8f;
typedef __attribute__((ext_vector_type(4)))  int    v4i;

#if defined(__has_builtin)
#if __has_builtin(__builtin_amdgcn_global_load_async_to_lds_b128) && \
    __has_builtin(__builtin_amdgcn_s_wait_asynccnt)
#define USE_ASYNC_LDS 1
#endif
#endif
#ifndef USE_ASYNC_LDS
#define USE_ASYNC_LDS 0
#endif

#if USE_ASYNC_LDS
typedef __attribute__((address_space(1))) v4i* gptr_v4i;   // global (AS1)
typedef __attribute__((address_space(3))) v4i* lptr_v4i;   // LDS (AS3)
#endif

namespace {
constexpr int kB    = 2;
constexpr int kS    = 2048;
constexpr int kD    = 1024;
constexpr int kH    = 16;
constexpr int kDh   = 64;
constexpr int kBS   = kB * kS;        // 4096
constexpr int kNQKV = 3 * kH * kDh;   // 3072
}

__device__ __forceinline__ v8f wmma_bf16(v16bf a, v16bf b, v8f c) {
  return __builtin_amdgcn_wmma_f32_16x16x32_bf16(
      /*neg_a=*/false, a, /*neg_b=*/false, b,
      /*c_mod=*/(short)0, c, /*reuse_a=*/false, /*reuse_b=*/false);
}

// ---- global -> LDS staging helpers ------------------------------------------
// Copy one 16-byte chunk per lane. Async path: no VGPR round trip (ASYNCcnt).
__device__ __forceinline__ void stage16(const bf16_t* g, bf16_t* l) {
#if USE_ASYNC_LDS
  __builtin_amdgcn_global_load_async_to_lds_b128(
      (gptr_v4i)(g), (lptr_v4i)(l), 0, 0);
#else
  *(v8bf*)l = *(const v8bf*)g;
#endif
}

__device__ __forceinline__ void stage_wait() {
#if USE_ASYNC_LDS
  __builtin_amdgcn_s_wait_asynccnt(0);
#endif
}

// Two 32-byte chunks from different streams, loads batched ahead of stores.
__device__ __forceinline__ void stage32x2(const bf16_t* ga, bf16_t* la,
                                          const bf16_t* gb, bf16_t* lb) {
#if USE_ASYNC_LDS
  stage16(ga, la);
  stage16(ga + 8, la + 8);
  stage16(gb, lb);
  stage16(gb + 8, lb + 8);
#else
  v8bf a0 = *(const v8bf*)(ga);
  v8bf a1 = *(const v8bf*)(ga + 8);
  v8bf b0 = *(const v8bf*)(gb);
  v8bf b1 = *(const v8bf*)(gb + 8);
  *(v8bf*)(la)     = a0;
  *(v8bf*)(la + 8) = a1;
  *(v8bf*)(lb)     = b0;
  *(v8bf*)(lb + 8) = b1;
#endif
}

// A-operand fragment (16x32 bf16, row-major source, stride in elements).
// Lane m=lane&15; halves 0..7 = K[kb..kb+7], halves 8..15 = K[kb+16..kb+23], kb=8*(lane>>4).
__device__ __forceinline__ v16bf load_frag_a(const bf16_t* base, int stride) {
  const int lane = threadIdx.x & 31;
  const int r  = lane & 15;
  const int kb = (lane >> 4) << 3;
  const bf16_t* p = base + r * stride + kb;
  union { v16bf f; v8bf h[2]; } u;
  u.h[0] = *(const v8bf*)(p);
  u.h[1] = *(const v8bf*)(p + 16);
  return u.f;
}

// B-operand fragment (32x16 bf16) from B^T row-major (column-major B).
// Lane n=lane&15 holds column n; halves 0..7 = K[kb..kb+7], halves 8..15 = K[kb+16..kb+23], kb=8*(lane>>4).
__device__ __forceinline__ v16bf load_frag_b(const bf16_t* base, int stride) {
  const int lane = threadIdx.x & 31;
  const int n  = lane & 15;
  const int kb = (lane >> 4) << 3;
  const bf16_t* p = base + n * stride + kb;
  union { v16bf f; v8bf h[2]; } u;
  u.h[0] = *(const v8bf*)(p);
  u.h[1] = *(const v8bf*)(p + 16);
  return u.f;
}

// ---------------- Kernel 1: convert / transpose into bf16 workspace ----------
__global__ void mha_prep(const float* __restrict__ x,
                         const float* __restrict__ wq, const float* __restrict__ wk,
                         const float* __restrict__ wv, const float* __restrict__ wo,
                         bf16_t* __restrict__ xb, bf16_t* __restrict__ wqkvt,
                         bf16_t* __restrict__ wot) {
  const long nx = (long)kBS * kD;
  const long nw = (long)kNQKV * kD;
  const long no = (long)kD * kD;
  const long total = nx + nw + no;
  for (long i = (long)blockIdx.x * blockDim.x + threadIdx.x; i < total;
       i += (long)gridDim.x * blockDim.x) {
    if (i < nx) {
      xb[i] = (bf16_t)x[i];
    } else if (i < nx + nw) {
      long j = i - nx;
      int n = (int)(j >> 10);        // packed output column 0..3071
      int d = (int)(j & 1023);       // K index
      int p = n >> 10;               // 0=q 1=k 2=v
      int h = (n >> 6) & 15;
      int e = n & 63;
      const float* w = (p == 0) ? wq : (p == 1) ? wk : wv;   // [H][D][Dh]
      wqkvt[j] = (bf16_t)w[((long)h * kD + d) * kDh + e];
    } else {
      long j = i - nx - nw;
      int e = (int)(j >> 10);
      int d = (int)(j & 1023);
      wot[j] = (bf16_t)wo[(long)d * kD + e];                 // Wo^T
    }
  }
}

// ---------------- Kernel 2: fused QKV projection GEMM ------------------------
// C[4096 x 3072] = xb @ Wqkv, bias added; Q scaled by 1/sqrt(Dh); V stored transposed.
__global__ __launch_bounds__(256)
void mha_qkv(const bf16_t* __restrict__ xb, const bf16_t* __restrict__ wqkvt,
             const float* __restrict__ qbias, const float* __restrict__ kbias,
             const float* __restrict__ vbias,
             bf16_t* __restrict__ qbuf, bf16_t* __restrict__ kbuf,
             bf16_t* __restrict__ vtbuf) {
  __shared__ __align__(16) bf16_t lds_a[128 * 32];
  __shared__ __align__(16) bf16_t lds_b[128 * 32];

  const int m0 = blockIdx.x * 128;
  const int n0 = blockIdx.y * 128;
  const int t  = threadIdx.x;
  const int w  = t >> 5;
  const int wm = w >> 1;                 // 0..3 -> 32-row strip
  const int wn = w & 1;                  // 0..1 -> 64-col strip
  const int lane = t & 31;
  const int half = lane >> 4;
  const int cn   = lane & 15;

  const v8f vzero = {};
  v8f acc[2][4];
#pragma unroll
  for (int i = 0; i < 2; ++i)
#pragma unroll
    for (int j = 0; j < 4; ++j) acc[i][j] = vzero;

  const int rowT = t >> 1;               // 0..127
  const int colT = (t & 1) * 16;         // 0 / 16

  for (int k0 = 0; k0 < kD; k0 += 32) {
    const bf16_t* ga = xb    + (long)(m0 + rowT) * kD + k0 + colT;
    const bf16_t* gb = wqkvt + (long)(n0 + rowT) * kD + k0 + colT;
    stage32x2(ga, lds_a + rowT * 32 + colT, gb, lds_b + rowT * 32 + colT);
    if (k0 + 32 < kD) {                  // -> global_prefetch_b8
      __builtin_prefetch(ga + 32, 0, 3);
      __builtin_prefetch(gb + 32, 0, 3);
    }
    stage_wait();
    __syncthreads();

    v16bf af[2], bf[4];
#pragma unroll
    for (int i = 0; i < 2; ++i)
      af[i] = load_frag_a(lds_a + (wm * 32 + i * 16) * 32, 32);
#pragma unroll
    for (int j = 0; j < 4; ++j)
      bf[j] = load_frag_b(lds_b + (wn * 64 + j * 16) * 32, 32);
#pragma unroll
    for (int i = 0; i < 2; ++i)
#pragma unroll
      for (int j = 0; j < 4; ++j)
        acc[i][j] = wmma_bf16(af[i], bf[j], acc[i][j]);
    __syncthreads();
  }

  // Epilogue: bias, scale, scatter to q/k/vT layouts.
#pragma unroll
  for (int j = 0; j < 4; ++j) {
    const int n = n0 + wn * 64 + j * 16 + cn;
    const int p = n >> 10;
    const int h = (n >> 6) & 15;
    const int e = n & 63;
    const float* bp = (p == 0) ? qbias : (p == 1) ? kbias : vbias;
    const float bias = bp[h * kDh + e];
#pragma unroll
    for (int i = 0; i < 2; ++i) {
#pragma unroll
      for (int r = 0; r < 8; ++r) {
        const int m = m0 + wm * 32 + i * 16 + r + 8 * half;
        const int b = m >> 11;
        const int s = m & (kS - 1);
        const float val = acc[i][j][r] + bias;
        if (p == 0) {
          qbuf[((long)(b * kH + h) * kS + s) * kDh + e] = (bf16_t)(val * 0.125f);
        } else if (p == 1) {
          kbuf[((long)(b * kH + h) * kS + s) * kDh + e] = (bf16_t)val;
        } else {
          vtbuf[((long)(b * kH + h) * kDh + e) * kS + s] = (bf16_t)val;  // V^T
        }
      }
    }
  }
}

// ---------------- Kernel 3: flash attention per (b, h, 128-q-row block) ------
__global__ __launch_bounds__(256)
void mha_attn(const bf16_t* __restrict__ qbuf, const bf16_t* __restrict__ kbuf,
              const bf16_t* __restrict__ vtbuf, const unsigned char* __restrict__ mask,
              bf16_t* __restrict__ attnb) {
  __shared__ __align__(16) bf16_t lds_k[64 * 64];
  __shared__ __align__(16) bf16_t lds_v[64 * 64];
  __shared__ __align__(16) bf16_t lds_p[8 * 16 * 64];

  const int bid  = blockIdx.x;            // 512 blocks
  const int qblk = bid & 15;
  const int h    = (bid >> 4) & 15;
  const int b    = bid >> 8;
  const int t    = threadIdx.x;
  const int w    = t >> 5;
  const int lane = t & 31;
  const int half = lane >> 4;
  const int cn   = lane & 15;

  const bf16_t* qh = qbuf  + (long)(b * kH + h) * kS * kDh;
  const bf16_t* kh = kbuf  + (long)(b * kH + h) * kS * kDh;
  const bf16_t* vh = vtbuf + (long)(b * kH + h) * kDh * kS;
  const unsigned char* mk = mask + (long)b * kS;

  const int q0 = qblk * 128 + w * 16;     // 16 query rows per wave
  const v16bf aq0 = load_frag_a(qh + (long)q0 * kDh, kDh);
  const v16bf aq1 = load_frag_a(qh + (long)q0 * kDh + 32, kDh);

  const v8f vzero = {};
  float mrow[8], lsum[8];
  v8f o[4];
#pragma unroll
  for (int r = 0; r < 8; ++r) { mrow[r] = -3.0e38f; lsum[r] = 0.0f; }
#pragma unroll
  for (int j = 0; j < 4; ++j) o[j] = vzero;

  const int rowT = t >> 2;                // 0..63
  const int colT = (t & 3) * 16;

  for (int j0 = 0; j0 < kS; j0 += 64) {
    // stage K tile (rows = keys) and V^T tile (rows = e) in LDS
    stage32x2(kh + (long)(j0 + rowT) * kDh + colT, lds_k + rowT * 64 + colT,
              vh + (long)rowT * kS + j0 + colT,    lds_v + rowT * 64 + colT);
    if (j0 + 64 < kS) {
      __builtin_prefetch(kh + (long)(j0 + 64 + rowT) * kDh + colT, 0, 3);
      __builtin_prefetch(vh + (long)rowT * kS + j0 + 64 + colT, 0, 3);
    }
    stage_wait();
    __syncthreads();

    // S = Q K^T for 4 key subtiles (Q pre-scaled by 1/sqrt(Dh))
    v8f sc[4];
#pragma unroll
    for (int ns = 0; ns < 4; ++ns) {
      v16bf bk0 = load_frag_b(lds_k + ns * 16 * 64, 64);
      v16bf bk1 = load_frag_b(lds_k + ns * 16 * 64 + 32, 64);
      v8f a = vzero;
      a = wmma_bf16(aq0, bk0, a);
      a = wmma_bf16(aq1, bk1, a);
      const int kidx = j0 + ns * 16 + cn;
      if (!mk[kidx]) {
#pragma unroll
        for (int r = 0; r < 8; ++r) a[r] = -3.0e38f;
      }
      sc[ns] = a;
    }

    // online softmax (rows live across 16-lane halves -> xor butterflies)
    float corr[8];
#pragma unroll
    for (int r = 0; r < 8; ++r) {
      float mx = fmaxf(fmaxf(sc[0][r], sc[1][r]), fmaxf(sc[2][r], sc[3][r]));
      for (int d = 1; d <= 8; d <<= 1) mx = fmaxf(mx, __shfl_xor(mx, d, 32));
      const float mnew = fmaxf(mrow[r], mx);
      corr[r] = __expf(mrow[r] - mnew);
      mrow[r] = mnew;
    }
    float rs[8];
#pragma unroll
    for (int r = 0; r < 8; ++r) rs[r] = 0.0f;
#pragma unroll
    for (int ns = 0; ns < 4; ++ns)
#pragma unroll
      for (int r = 0; r < 8; ++r) {
        const float p = __expf(sc[ns][r] - mrow[r]);
        sc[ns][r] = p;
        rs[r] += p;
      }
#pragma unroll
    for (int r = 0; r < 8; ++r) {
      for (int d = 1; d <= 8; d <<= 1) rs[r] += __shfl_xor(rs[r], d, 32);
      lsum[r] = lsum[r] * corr[r] + rs[r];
    }
#pragma unroll
    for (int j = 0; j < 4; ++j)
#pragma unroll
      for (int r = 0; r < 8; ++r) o[j][r] *= corr[r];

    // C-layout -> A-layout for P via wave-private LDS slab (DS in-order per wave)
    bf16_t* pw = lds_p + w * 16 * 64;
#pragma unroll
    for (int ns = 0; ns < 4; ++ns)
#pragma unroll
      for (int r = 0; r < 8; ++r)
        pw[(r + 8 * half) * 64 + ns * 16 + cn] = (bf16_t)sc[ns][r];

    const v16bf pa0 = load_frag_a(pw, 64);
    const v16bf pa1 = load_frag_a(pw + 32, 64);

    // O += P V  (B^T = V^T rows are contiguous in lds_v)
#pragma unroll
    for (int es = 0; es < 4; ++es) {
      v16bf bv0 = load_frag_b(lds_v + es * 16 * 64, 64);
      v16bf bv1 = load_frag_b(lds_v + es * 16 * 64 + 32, 64);
      o[es] = wmma_bf16(pa0, bv0, o[es]);
      o[es] = wmma_bf16(pa1, bv1, o[es]);
    }
    __syncthreads();
  }

  // normalize and store attn output [BS x H*Dh] bf16 (A matrix of final GEMM)
#pragma unroll
  for (int es = 0; es < 4; ++es)
#pragma unroll
    for (int r = 0; r < 8; ++r) {
      const int qrow = q0 + r + 8 * half;
      const int col  = h * kDh + es * 16 + cn;
      attnb[((long)(b * kS + qrow)) * (kH * kDh) + col] = (bf16_t)(o[es][r] / lsum[r]);
    }
}

// ---------------- Kernel 4: output projection (f32 result) -------------------
__global__ __launch_bounds__(256)
void mha_out(const bf16_t* __restrict__ attnb, const bf16_t* __restrict__ wot,
             float* __restrict__ out) {
  __shared__ __align__(16) bf16_t lds_a[128 * 32];
  __shared__ __align__(16) bf16_t lds_b[128 * 32];

  const int m0 = blockIdx.x * 128;
  const int n0 = blockIdx.y * 128;
  const int t  = threadIdx.x;
  const int w  = t >> 5;
  const int wm = w >> 1;
  const int wn = w & 1;
  const int lane = t & 31;
  const int half = lane >> 4;
  const int cn   = lane & 15;

  const v8f vzero = {};
  v8f acc[2][4];
#pragma unroll
  for (int i = 0; i < 2; ++i)
#pragma unroll
    for (int j = 0; j < 4; ++j) acc[i][j] = vzero;

  const int rowT = t >> 1;
  const int colT = (t & 1) * 16;

  for (int k0 = 0; k0 < kD; k0 += 32) {
    const bf16_t* ga = attnb + (long)(m0 + rowT) * kD + k0 + colT;
    const bf16_t* gb = wot   + (long)(n0 + rowT) * kD + k0 + colT;
    stage32x2(ga, lds_a + rowT * 32 + colT, gb, lds_b + rowT * 32 + colT);
    if (k0 + 32 < kD) {
      __builtin_prefetch(ga + 32, 0, 3);
      __builtin_prefetch(gb + 32, 0, 3);
    }
    stage_wait();
    __syncthreads();

    v16bf af[2], bf[4];
#pragma unroll
    for (int i = 0; i < 2; ++i)
      af[i] = load_frag_a(lds_a + (wm * 32 + i * 16) * 32, 32);
#pragma unroll
    for (int j = 0; j < 4; ++j)
      bf[j] = load_frag_b(lds_b + (wn * 64 + j * 16) * 32, 32);
#pragma unroll
    for (int i = 0; i < 2; ++i)
#pragma unroll
      for (int j = 0; j < 4; ++j)
        acc[i][j] = wmma_bf16(af[i], bf[j], acc[i][j]);
    __syncthreads();
  }

#pragma unroll
  for (int i = 0; i < 2; ++i)
#pragma unroll
    for (int j = 0; j < 4; ++j)
#pragma unroll
      for (int r = 0; r < 8; ++r) {
        const int m = m0 + wm * 32 + i * 16 + r + 8 * half;
        const int n = n0 + wn * 64 + j * 16 + cn;
        out[(long)m * kD + n] = acc[i][j][r];
      }
}

// ---------------- Host launcher ----------------------------------------------
extern "C" void kernel_launch(void* const* d_in, const int* in_sizes, int n_in,
                              void* d_out, int out_size, void* d_ws, size_t ws_size,
                              hipStream_t stream) {
  const float* x            = (const float*)d_in[0];
  const unsigned char* mask = (const unsigned char*)d_in[1];   // jnp bool = 1 byte
  const float* wq           = (const float*)d_in[2];
  const float* qb           = (const float*)d_in[3];
  const float* wk           = (const float*)d_in[4];
  const float* kb           = (const float*)d_in[5];
  const float* wv           = (const float*)d_in[6];
  const float* vb           = (const float*)d_in[7];
  const float* wo           = (const float*)d_in[8];
  float* out                = (float*)d_out;

  bf16_t* ws    = (bf16_t*)d_ws;
  bf16_t* xb    = ws;                                   // 4096*1024
  bf16_t* wqkvt = xb    + (size_t)kBS * kD;             // 3072*1024
  bf16_t* wot   = wqkvt + (size_t)kNQKV * kD;           // 1024*1024
  bf16_t* qbuf  = wot   + (size_t)kD * kD;              // B*H*S*Dh
  bf16_t* kbuf  = qbuf  + (size_t)kB * kH * kS * kDh;
  bf16_t* vtbuf = kbuf  + (size_t)kB * kH * kS * kDh;
  bf16_t* attnb = vtbuf + (size_t)kB * kH * kS * kDh;   // 4096*1024

  mha_prep<<<8192, 256, 0, stream>>>(x, wq, wk, wv, wo, xb, wqkvt, wot);
  mha_qkv<<<dim3(kBS / 128, kNQKV / 128), 256, 0, stream>>>(
      xb, wqkvt, qb, kb, vb, qbuf, kbuf, vtbuf);
  mha_attn<<<kB * kH * (kS / 128), 256, 0, stream>>>(qbuf, kbuf, vtbuf, mask, attnb);
  mha_out<<<dim3(kBS / 128, kD / 128), 256, 0, stream>>>(attnb, wot, out);
}
